// PainterNet_85366769975390
// MI455X (gfx1250) — matrix operand order repaired and checked by reading, not compiled, not run on hardware
//
#include <hip/hip_runtime.h>
#include <hip/hip_bf16.h>

// ---------------------------------------------------------------------------
// Types for CDNA5 WMMA
// ---------------------------------------------------------------------------
typedef __attribute__((ext_vector_type(16))) __bf16        v16bf;
typedef __attribute__((ext_vector_type(8)))  float         v8f;
typedef __attribute__((ext_vector_type(4)))  unsigned int  u32x4;

union Frag {
    u32x4 u[2];
    v16bf v;
};

__device__ __forceinline__ v8f vzero8() {
    v8f z;
#pragma unroll
    for (int i = 0; i < 8; ++i) z[i] = 0.0f;
    return z;
}

__device__ __forceinline__ unsigned short f2bf(float x) {
    unsigned int u = __float_as_uint(x);
    unsigned int r = (u + 0x7FFFu + ((u >> 16) & 1u)) >> 16;   // round-to-nearest-even
    return (unsigned short)r;
}

__device__ __forceinline__ int iclamp(int v, int lo, int hi) {
    return v < lo ? lo : (v > hi ? hi : v);
}

// CDNA5 async global->LDS staging (GVS mode: saddr base + 32-bit voffset).
// vdst VGPR holds the LDS byte address (low 32 bits of the generic pointer).
__device__ __forceinline__ void async_ld128(const unsigned short* ldsdst, unsigned voff,
                                            unsigned long long base) {
    unsigned ldsaddr = (unsigned)(size_t)ldsdst;
    asm volatile("global_load_async_to_lds_b128 %0, %1, %2"
                 :: "v"(ldsaddr), "v"(voff), "s"(base) : "memory");
}

__device__ __forceinline__ void wait_async0() {
    asm volatile("s_wait_asynccnt 0x0" ::: "memory");
}

// Problem constants
#define CCH   128
#define HH    128
#define HR    64
#define LL    4096           // HR*HR
#define K1    1152           // CCH*9
#define SSCALE 10.0f

// ---------------------------------------------------------------------------
// Prep: Bm[p][k] = zero-padded 3x3 im2col of f[::2,::2]   (bf16, p-major, K1 cols)
// ---------------------------------------------------------------------------
__global__ void k_build_Bm(const float* __restrict__ f, unsigned short* __restrict__ Bm) {
    int idx = blockIdx.x * blockDim.x + threadIdx.x;
    if (idx >= LL * K1) return;
    int k = idx % K1;
    int p = idx / K1;
    int y = p / HR, x = p % HR;
    int cc = k / 9;
    int r  = k % 9;
    int i = r / 3, j = r % 3;
    int yy = y + i - 1, xx = x + j - 1;
    float v = 0.0f;
    if (yy >= 0 && yy < HR && xx >= 0 && xx < HR)
        v = f[((size_t)cc * HH + 2 * yy) * HH + 2 * xx];
    Bm[idx] = f2bf(v);
}

// ---------------------------------------------------------------------------
// Prep: A[l][k] = L2-normalized low-res bg patch (edge-clamped), and mi[l]
// ---------------------------------------------------------------------------
__global__ __launch_bounds__(128) void k_build_A(const float* __restrict__ b,
                                                 const float* __restrict__ mask,
                                                 unsigned short* __restrict__ A,
                                                 float* __restrict__ mi) {
    int l  = blockIdx.x;           // 0..4095
    int ph = l / HR, pw = l % HR;
    int cc = threadIdx.x;          // 0..127 == channel

    float vals[9];
    float ss = 0.0f;
#pragma unroll
    for (int e = 0; e < 9; ++e) {
        int i = e / 3, j = e % 3;
        int yy = iclamp(ph + i - 1, 0, HR - 1);
        int xx = iclamp(pw + j - 1, 0, HR - 1);
        float v = b[((size_t)cc * HH + 2 * yy) * HH + 2 * xx];
        vals[e] = v;
        ss += v * v;
    }
    __shared__ float red[128];
    red[cc] = ss;
    __syncthreads();
    for (int s = 64; s > 0; s >>= 1) {
        if (cc < s) red[cc] += red[cc + s];
        __syncthreads();
    }
    float bn  = fmaxf(sqrtf(red[0]), 1e-4f);
    float inv = 1.0f / bn;
#pragma unroll
    for (int e = 0; e < 9; ++e)
        A[(size_t)l * K1 + cc * 9 + e] = f2bf(vals[e] * inv);

    if (cc == 0) {
        float s = 0.0f;
#pragma unroll
        for (int e = 0; e < 9; ++e) {
            int i = e / 3, j = e % 3;
            int yy = ph + i - 1, xx = pw + j - 1;
            if (yy >= 0 && yy < HR && xx >= 0 && xx < HR)
                s += mask[(size_t)(2 * yy) * HH + 2 * xx];
        }
        mi[l] = (s == 0.0f) ? 1.0f : 0.0f;
    }
}

// ---------------------------------------------------------------------------
// Prep: BGT[k'][l] = hi-res bg patch element (edge-clamped), transposed layout
// ---------------------------------------------------------------------------
__global__ void k_build_BGT(const float* __restrict__ b, unsigned short* __restrict__ BGT) {
    int idx = blockIdx.x * blockDim.x + threadIdx.x;
    if (idx >= K1 * LL) return;
    int l = idx % LL;
    int k = idx / LL;
    int ph = l / HR, pw = l % HR;
    int cc = k / 9;
    int r  = k % 9;
    int u = r / 3, v = r % 3;
    int Yy = iclamp(2 * ph + u - 1, 0, HH - 1);
    int Xx = iclamp(2 * pw + v - 1, 0, HH - 1);
    BGT[idx] = f2bf(b[((size_t)cc * HH + Yy) * HH + Xx]);
}

// ---------------------------------------------------------------------------
// Generic bf16 GEMM:  O[M x N] (f32) = X[M x K] * Y[N x K]^T   via WMMA bf16
//   block tile 128x128, BK = 32, 256 threads = 8 waves, each wave 64x32.
//   Double-buffered LDS staged with global_load_async_to_lds_b128 (ASYNCcnt).
// ---------------------------------------------------------------------------
#define LDSPITCH 40   // ushort pitch (32 data + 8 pad), keeps 16B alignment

__global__ __launch_bounds__(256) void gemm_bf16_xyt(const unsigned short* __restrict__ X,
                                                     const unsigned short* __restrict__ Y,
                                                     float* __restrict__ O,
                                                     int M, int N, int K) {
    __shared__ __align__(16) unsigned short Xs[2][128 * LDSPITCH];
    __shared__ __align__(16) unsigned short Ys[2][128 * LDSPITCH];

    const int tid  = threadIdx.x;
    const int wid  = tid >> 5;
    const int lane = tid & 31;
    const int n0 = blockIdx.x * 128;
    const int m0 = blockIdx.y * 128;
    const int wm = (wid >> 2) * 64;   // wave M offset within tile
    const int wn = (wid & 3) * 32;    // wave N offset within tile
    const int lr = lane & 15;         // row within 16
    const int kh = lane >> 4;         // k-half select

    const unsigned long long Xb64 = (unsigned long long)(size_t)X;
    const unsigned long long Yb64 = (unsigned long long)(size_t)Y;
    const int row = tid >> 1;         // 0..127: this thread's staging row
    const int kk0 = (tid & 1) * 16;   // first 8-element chunk column (covers 2x b128)

    v8f acc[4][2];
#pragma unroll
    for (int i = 0; i < 4; ++i)
#pragma unroll
        for (int j = 0; j < 2; ++j) acc[i][j] = vzero8();

    // Each thread asynchronously stages 2x b128 of X and 2x b128 of Y per tile:
    // rows = tid>>1, columns [kk0, kk0+8) and [kk0+8? ...] -> chunks kk0 and kk0+8.
    auto stage = [&](int buf, int k0) {
#pragma unroll
        for (int c = 0; c < 2; ++c) {
            int kk = kk0 + c * 8;     // 0,8 or 16,24
            async_ld128(&Xs[buf][row * LDSPITCH + kk],
                        (unsigned)(((size_t)(m0 + row) * K + k0 + kk) * 2), Xb64);
            async_ld128(&Ys[buf][row * LDSPITCH + kk],
                        (unsigned)(((size_t)(n0 + row) * K + k0 + kk) * 2), Yb64);
        }
    };

    const int nk = K >> 5;
    stage(0, 0);                       // prologue: tile 0 -> buffer 0

    for (int t = 0; t < nk; ++t) {
        wait_async0();                 // my async stores for tile t have landed in LDS
        __syncthreads();               // everyone's have; prev-tile readers are done

        if (t + 1 < nk)                // overlap: stage tile t+1 while computing t
            stage((t + 1) & 1, (t + 1) << 5);
        if (t + 2 < nk) {              // warm L2 for tile t+2 (global_prefetch_b8)
            __builtin_prefetch(X + (size_t)(m0 + row) * K + ((t + 2) << 5), 0, 1);
            __builtin_prefetch(Y + (size_t)(n0 + row) * K + ((t + 2) << 5), 0, 1);
        }

        const unsigned short* Xb = Xs[t & 1];
        const unsigned short* Yb = Ys[t & 1];

        // ---- load fragments per ISA VGPR layout, issue 8 WMMAs ----
        Frag a[4], bfr[2];
#pragma unroll
        for (int s = 0; s < 4; ++s) {
            const unsigned short* p = &Xb[(wm + s * 16 + lr) * LDSPITCH + kh * 8];
            a[s].u[0] = *(const u32x4*)p;          // K = kh*8 .. +7
            a[s].u[1] = *(const u32x4*)(p + 16);   // K = 16+kh*8 .. +7
        }
#pragma unroll
        for (int s = 0; s < 2; ++s) {
            const unsigned short* p = &Yb[(wn + s * 16 + lr) * LDSPITCH + kh * 8];
            bfr[s].u[0] = *(const u32x4*)p;        // K = kh*8 .. +7
            bfr[s].u[1] = *(const u32x4*)(p + 16); // K = 16+kh*8 .. +7
        }
#pragma unroll
        for (int i = 0; i < 4; ++i)
#pragma unroll
            for (int j = 0; j < 2; ++j)
                acc[i][j] = __builtin_amdgcn_wmma_f32_16x16x32_bf16(
                    false, a[i].v, false, bfr[j].v, (short)0, acc[i][j], false, false);
    }

    // ---- store D (f32 16x16 layout: M = r + 8*kh, N = lane&15) ----
#pragma unroll
    for (int i = 0; i < 4; ++i)
#pragma unroll
        for (int j = 0; j < 2; ++j)
#pragma unroll
            for (int r = 0; r < 8; ++r) {
                int m = m0 + wm + i * 16 + r + 8 * kh;
                int n = n0 + wn + j * 16 + lr;
                O[(size_t)m * N + n] = acc[i][j][r];
            }
}

// ---------------------------------------------------------------------------
// Fused: fuse_diag -> transpose -> fuse_diag -> transpose-back, on scoreT[p][l]
//   S0[a=l, b=p] lives at S[p*4096 + l]; swap(x) = (x%64)*64 + x/64
// ---------------------------------------------------------------------------
__global__ void k_fuse(const float* __restrict__ S, float* __restrict__ F) {
    int idx = blockIdx.x * blockDim.x + threadIdx.x;
    if (idx >= LL * LL) return;
    int l = idx & (LL - 1);
    int p = idx >> 12;
    int ap = ((l & 63) << 6) | (l >> 6);
    int bp = ((p & 63) << 6) | (p >> 6);
    float sum = 0.0f;
#pragma unroll
    for (int t2 = -1; t2 <= 1; ++t2) {
        int aa = ap + t2, bb = bp + t2;
        if (aa < 0 || aa >= LL || bb < 0 || bb >= LL) continue;
        int Aq = ((aa & 63) << 6) | (aa >> 6);
        int Bq = ((bb & 63) << 6) | (bb >> 6);
#pragma unroll
        for (int t1 = -1; t1 <= 1; ++t1) {
            int a1 = Aq + t1, b1 = Bq + t1;
            if (a1 < 0 || a1 >= LL || b1 < 0 || b1 >= LL) continue;
            sum += S[((size_t)b1 << 12) + a1];
        }
    }
    F[idx] = sum;
}

// ---------------------------------------------------------------------------
// Softmax over l (rows of F[p][.]), gated by mi; writes bf16 attn[p][l]
// ---------------------------------------------------------------------------
__global__ __launch_bounds__(256) void k_softmax(const float* __restrict__ F,
                                                 const float* __restrict__ mi,
                                                 unsigned short* __restrict__ attn) {
    int p = blockIdx.x;
    int t = threadIdx.x;
    __shared__ float red[256];
    const float* row = F + ((size_t)p << 12);

    float mx = -1e30f;
    for (int l = t; l < LL; l += 256)
        mx = fmaxf(mx, SSCALE * row[l] * mi[l]);
    red[t] = mx;
    __syncthreads();
    for (int s = 128; s > 0; s >>= 1) {
        if (t < s) red[t] = fmaxf(red[t], red[t + s]);
        __syncthreads();
    }
    mx = red[0];
    __syncthreads();

    float sum = 0.0f;
    for (int l = t; l < LL; l += 256)
        sum += __expf(SSCALE * row[l] * mi[l] - mx);
    red[t] = sum;
    __syncthreads();
    for (int s = 128; s > 0; s >>= 1) {
        if (t < s) red[t] += red[t + s];
        __syncthreads();
    }
    float inv = 1.0f / red[0];
    for (int l = t; l < LL; l += 256) {
        float w = __expf(SSCALE * row[l] * mi[l] - mx) * mi[l] * inv;
        attn[((size_t)p << 12) + l] = f2bf(w);
    }
}

// ---------------------------------------------------------------------------
// col2im: stride-2 overlap-add of out_patches[p][c*9+u*3+v], /4, edge-pad 127->128
// ---------------------------------------------------------------------------
__global__ void k_col2im(const float* __restrict__ OP, float* __restrict__ out) {
    int idx = blockIdx.x * blockDim.x + threadIdx.x;
    if (idx >= CCH * HH * HH) return;
    int Xp = idx % HH;
    int Yp = (idx / HH) % HH;
    int cc = idx / (HH * HH);
    int Yc = Yp < 127 ? Yp : 126;   // edge pad replicates row/col 126
    int Xc = Xp < 127 ? Xp : 126;
    float sum = 0.0f;
#pragma unroll
    for (int u = 0; u < 3; ++u) {
        int y2 = Yc + 1 - u;
        if (y2 < 0 || (y2 & 1)) continue;
        int y = y2 >> 1;
        if (y >= HR) continue;
#pragma unroll
        for (int v = 0; v < 3; ++v) {
            int x2 = Xc + 1 - v;
            if (x2 < 0 || (x2 & 1)) continue;
            int x = x2 >> 1;
            if (x >= HR) continue;
            sum += OP[(size_t)(y * HR + x) * K1 + cc * 9 + u * 3 + v];
        }
    }
    out[idx] = sum * 0.25f;
}

// ---------------------------------------------------------------------------
// Host launcher
// ---------------------------------------------------------------------------
extern "C" void kernel_launch(void* const* d_in, const int* in_sizes, int n_in,
                              void* d_out, int out_size, void* d_ws, size_t ws_size,
                              hipStream_t stream) {
    (void)in_sizes; (void)n_in; (void)out_size; (void)ws_size;
    const float* f_all    = (const float*)d_in[0];   // (4,128,128,128)
    const float* b_all    = (const float*)d_in[1];   // (4,128,128,128)
    const float* mask_all = (const float*)d_in[2];   // (4,1,128,128)
    float*       out_all  = (float*)d_out;           // (4,128,128,128)

    // workspace carve-up (reused per batch item; kernels serialize on stream)
    char* ws = (char*)d_ws;
    size_t off = 0;
    auto carve = [&](size_t bytes) { size_t r = off; off = (off + bytes + 255) & ~(size_t)255; return r; };
    unsigned short* A   = (unsigned short*)(ws + carve((size_t)LL * K1 * 2));   // bgs_n
    unsigned short* Bm  = (unsigned short*)(ws + carve((size_t)LL * K1 * 2));   // fp im2col
    unsigned short* BGT = (unsigned short*)(ws + carve((size_t)K1 * LL * 2));   // bg^T
    float*          mi  = (float*)(ws + carve((size_t)LL * 4));
    float*          buf1 = (float*)(ws + carve((size_t)LL * LL * 4));           // scoreT / attn(bf16)
    float*          buf2 = (float*)(ws + carve((size_t)LL * LL * 4));           // fused
    float*          OP   = (float*)(ws + carve((size_t)LL * K1 * 4));           // out patches

    const size_t fbStride   = (size_t)CCH * HH * HH;   // per-batch f/b stride
    const size_t maskStride = (size_t)HH * HH;
    const size_t outStride  = (size_t)CCH * HH * HH;

    for (int bi = 0; bi < 4; ++bi) {
        const float* f = f_all + bi * fbStride;
        const float* b = b_all + bi * fbStride;
        const float* m = mask_all + bi * maskStride;
        float* out     = out_all + bi * outStride;

        k_build_Bm<<<(LL * K1 + 255) / 256, 256, 0, stream>>>(f, Bm);
        k_build_A<<<LL, 128, 0, stream>>>(b, m, A, mi);
        k_build_BGT<<<(K1 * LL + 255) / 256, 256, 0, stream>>>(b, BGT);

        // GEMM1: scoreT[p][l] = Bm[p][k] . A[l][k]
        gemm_bf16_xyt<<<dim3(LL / 128, LL / 128), 256, 0, stream>>>(Bm, A, buf1, LL, LL, K1);

        // fuse-transpose-fuse
        k_fuse<<<(LL * LL + 255) / 256, 256, 0, stream>>>(buf1, buf2);

        // softmax over l, write bf16 attention into buf1 region
        k_softmax<<<LL, 256, 0, stream>>>(buf2, mi, (unsigned short*)buf1);

        // GEMM2: OP[p][k'] = attn[p][l] . BGT[k'][l]
        gemm_bf16_xyt<<<dim3(K1 / 128, LL / 128), 256, 0, stream>>>(
            (const unsigned short*)buf1, BGT, OP, LL, K1, LL);

        // overlap-add + /4 + edge pad
        k_col2im<<<(CCH * HH * HH + 255) / 256, 256, 0, stream>>>(OP, out);
    }
}
